// TransformerDecoder_35923106464004
// MI455X (gfx1250) — matrix operand order repaired and checked by reading, not compiled, not run on hardware
//
#include <hip/hip_runtime.h>
#include <hip/hip_bf16.h>

typedef __attribute__((ext_vector_type(16))) __bf16 v16bf;
typedef __attribute__((ext_vector_type(8)))  float  v8f;
typedef unsigned int u32;
typedef __attribute__((ext_vector_type(4))) u32 u32x4;
typedef __attribute__((ext_vector_type(8))) int i32x8;
typedef __attribute__((ext_vector_type(4))) int i32x4;

union Frag { uint4 u[2]; v16bf v; };

#define DEPTH 4
#define HEADS 8
#define DHEAD 64
#define DIM   512
#define INNER 512
#define MLPD  2048
#define NTOK  1024
#define BATCH 8
#define MROWS (BATCH * NTOK)   // 8192
#define RED   32
#define SCALE 0.04419417382415922f   // 512^-0.5

#if __has_builtin(__builtin_amdgcn_tensor_load_to_lds) && __has_builtin(__builtin_amdgcn_s_wait_tensorcnt)
#define USE_TDM 1
#else
#define USE_TDM 0
#endif

#if USE_TDM
// Issue one TDM 2D tile load: tile (h rows x w elems of 2B), row stride in elements.
// Tensor dims == tile dims (tiles never go OOB here). One wave should execute this.
__device__ __forceinline__ void tdm_load_2d(unsigned lds_byte_off, const void* gptr,
                                            unsigned w, unsigned h,
                                            unsigned long long stride_elems) {
    unsigned long long ga = (unsigned long long)(size_t)gptr;
    u32x4 g0;
    g0.x = 1u;                                  // count=1, user mode
    g0.y = lds_byte_off;                        // LDS dest (bytes)
    g0.z = (u32)ga;                             // global_addr[31:0]
    g0.w = (u32)((ga >> 32) & 0x1ffffffu) | (2u << 30);   // addr[56:32] | type=2
    i32x8 g1;
    g1[0] = (int)(1u << 16);                                   // data_size=1 (2 bytes)
    g1[1] = (int)((w & 0xffffu) << 16);                        // tensor_dim0 lo
    g1[2] = (int)(((w >> 16) & 0xffffu) | ((h & 0xffffu) << 16));   // dim0 hi | dim1 lo
    g1[3] = (int)(((h >> 16) & 0xffffu) | ((w & 0xffffu) << 16));   // dim1 hi | tile_dim0
    g1[4] = (int)(h & 0xffffu);                                // tile_dim1 (tile_dim2=0)
    g1[5] = (int)(u32)(stride_elems & 0xffffffffu);            // dim0_stride lo32
    g1[6] = (int)(u32)((stride_elems >> 32) & 0xffffu);        // dim0_stride hi16
    g1[7] = 0;
    i32x4 z4 = {0, 0, 0, 0};
#if __clang_major__ >= 23
    i32x8 z8 = {0, 0, 0, 0, 0, 0, 0, 0};
    __builtin_amdgcn_tensor_load_to_lds(g0, g1, z4, z4, z8, 0);
#else
    __builtin_amdgcn_tensor_load_to_lds(g0, g1, z4, z4, 0);
#endif
}
#define TDM_WAIT() __builtin_amdgcn_s_wait_tensorcnt(0)
#else
#define TDM_WAIT()
#endif

// ---------------------------------------------------------------- utilities
__global__ void k_cvt(const float* __restrict__ s, __hip_bfloat16* __restrict__ d, long n) {
    long i = (long)blockIdx.x * blockDim.x + threadIdx.x;
    if (i < n) d[i] = __float2bfloat16(s[i]);
}

__global__ void k_copy(const float* __restrict__ s, float* __restrict__ d, long n) {
    long i = (long)blockIdx.x * blockDim.x + threadIdx.x;
    if (i < n) d[i] = s[i];
}

// ------------------------------------------------------------- layernorm -> bf16
__global__ __launch_bounds__(256) void k_ln(const float* __restrict__ x,
                                            const float* __restrict__ g,
                                            const float* __restrict__ bb,
                                            __hip_bfloat16* __restrict__ out) {
    int row = blockIdx.x, t = threadIdx.x;
    const float* xr = x + (size_t)row * DIM;
    float v0 = xr[t], v1 = xr[t + 256];
    float s = v0 + v1, q = v0 * v0 + v1 * v1;
#pragma unroll
    for (int off = 16; off > 0; off >>= 1) { s += __shfl_xor(s, off, 32); q += __shfl_xor(q, off, 32); }
    __shared__ float rs[8], rq[8], mu_s, is_s;
    int w = t >> 5, l = t & 31;
    if (l == 0) { rs[w] = s; rq[w] = q; }
    __syncthreads();
    if (t == 0) {
        float S = 0.f, Q = 0.f;
#pragma unroll
        for (int i = 0; i < 8; ++i) { S += rs[i]; Q += rq[i]; }
        float mu = S * (1.f / DIM);
        float var = Q * (1.f / DIM) - mu * mu;
        mu_s = mu; is_s = rsqrtf(var + 1e-5f);
    }
    __syncthreads();
    float mu = mu_s, inv = is_s;
    out[(size_t)row * DIM + t]       = __float2bfloat16((v0 - mu) * inv * g[t] + bb[t]);
    out[(size_t)row * DIM + t + 256] = __float2bfloat16((v1 - mu) * inv * g[t + 256] + bb[t + 256]);
}

// ------------------------------------------------------------- WMMA GEMM
// C[M,N] = A[M,K](bf16) * B[K,N](bf16) (+bias)(+gelu)(+resid) -> f32 or bf16
// FLAGS: 1=bias, 2=gelu, 4=resid(f32), 8=store f32 (else bf16)
#define BM 128
#define BN 64
#define BK 64
template <int FLAGS>
__global__ __launch_bounds__(256) void k_gemm(const __hip_bfloat16* __restrict__ A,
                                              const __hip_bfloat16* __restrict__ B,
                                              const float* __restrict__ bias,
                                              const float* __restrict__ resid,
                                              float* __restrict__ outF,
                                              __hip_bfloat16* __restrict__ outB,
                                              int M, int N, int K) {
    __shared__ __align__(16) __hip_bfloat16 As[2][BM][BK];   // [m][k]  2x16KB (TDM dest)
    __shared__ __align__(16) __hip_bfloat16 Bs[2][BN][BK];   // [n][k]  2x8KB (transposed)
    int t = threadIdx.x;
    int wave = t >> 5, lane = t & 31, l = lane & 15, hi = lane >> 4;
    int wm = wave & 3, wn = wave >> 2;
    int m0 = blockIdx.x * BM, n0 = blockIdx.y * BN;

    v8f zero = {0.f, 0.f, 0.f, 0.f, 0.f, 0.f, 0.f, 0.f};
    v8f acc[2][2];
#pragma unroll
    for (int a = 0; a < 2; ++a)
#pragma unroll
        for (int b = 0; b < 2; ++b) acc[a][b] = zero;

    auto loadA = [&](int buf, int k0) {
#if USE_TDM
        if (wave == 0)
            tdm_load_2d((unsigned)(size_t)&As[buf][0][0], A + (size_t)m0 * K + k0,
                        BK, BM, (unsigned long long)K);
#else
#pragma unroll
        for (int i = 0; i < 4; ++i) {
            int c = t * 4 + i, row = c >> 3, kc = (c & 7) * 8;
            *(uint4*)&As[buf][row][kc] = *(const uint4*)(A + (size_t)(m0 + row) * K + k0 + kc);
        }
#endif
    };
    auto loadB = [&](int buf, int k0) {
#pragma unroll
        for (int i = 0; i < 2; ++i) {
            int c = t * 2 + i;
            int k = c >> 3, n8 = (c & 7) * 8;
            uint4 bv = *(const uint4*)(B + (size_t)(k0 + k) * N + n0 + n8);
            const __hip_bfloat16* bp = (const __hip_bfloat16*)&bv;
#pragma unroll
            for (int j = 0; j < 8; ++j) Bs[buf][n8 + j][k] = bp[j];
        }
    };

    // prologue: stage 0
    loadA(0, 0);
    loadB(0, 0);
    TDM_WAIT();
    __syncthreads();

    int nsteps = K / BK;
    for (int s = 0; s < nsteps; ++s) {
        int cur = s & 1, nxt = cur ^ 1;
        if (s + 1 < nsteps) {               // prefetch next stage behind the WMMAs
            loadA(nxt, (s + 1) * BK);
            loadB(nxt, (s + 1) * BK);
        }
#pragma unroll
        for (int kc = 0; kc < 2; ++kc) {
            Frag af[2], bfr[2];
#pragma unroll
            for (int mt = 0; mt < 2; ++mt) {
                int row = wm * 32 + mt * 16 + l;
                af[mt].u[0] = *(const uint4*)&As[cur][row][kc * 32 + hi * 8];
                af[mt].u[1] = *(const uint4*)&As[cur][row][kc * 32 + 16 + hi * 8];
            }
#pragma unroll
            for (int nt = 0; nt < 2; ++nt) {
                int col = wn * 32 + nt * 16 + l;
                bfr[nt].u[0] = *(const uint4*)&Bs[cur][col][kc * 32 + hi * 8];
                bfr[nt].u[1] = *(const uint4*)&Bs[cur][col][kc * 32 + 16 + hi * 8];
            }
#pragma unroll
            for (int mt = 0; mt < 2; ++mt)
#pragma unroll
                for (int nt = 0; nt < 2; ++nt)
                    acc[mt][nt] = __builtin_amdgcn_wmma_f32_16x16x32_bf16(
                        false, af[mt].v, false, bfr[nt].v, (short)0, acc[mt][nt], false, false);
        }
        TDM_WAIT();
        __syncthreads();
    }
#pragma unroll
    for (int mt = 0; mt < 2; ++mt)
#pragma unroll
        for (int nt = 0; nt < 2; ++nt) {
            int col = n0 + wn * 32 + nt * 16 + l;
            float bval = (FLAGS & 1) ? bias[col] : 0.f;
#pragma unroll
            for (int r = 0; r < 8; ++r) {
                int row = m0 + wm * 32 + mt * 16 + hi * 8 + r;
                float v = acc[mt][nt][r] + bval;
                if (FLAGS & 2) v = 0.5f * v * (1.f + erff(v * 0.70710678118654752f));
                if (FLAGS & 4) v += resid[(size_t)row * N + col];
                if (FLAGS & 8) outF[(size_t)row * N + col] = v;
                else           outB[(size_t)row * N + col] = __float2bfloat16(v);
            }
        }
}

// ------------------------------------------------------------- flash attention
// grid.x = b*HEADS + h (64), grid.y = q tile of 64 rows (16). 4 waves.
__global__ __launch_bounds__(128) void k_attn(const __hip_bfloat16* __restrict__ Qg,
                                              const __hip_bfloat16* __restrict__ Kg,
                                              const __hip_bfloat16* __restrict__ Vg,
                                              __hip_bfloat16* __restrict__ Og,
                                              int ldq, int ldk, int ldv, float scale) {
    __shared__ __align__(16) __hip_bfloat16 Qs[64][64];
    __shared__ __align__(16) __hip_bfloat16 Ks[2][64][64];  // [j][d] == B layout [n][k]
    __shared__ __align__(16) __hip_bfloat16 Vs[2][64][64];  // [d][j] (transposed)
    __shared__ __align__(16) __hip_bfloat16 Ps[4][16][64];
    int t = threadIdx.x;
    int wave = t >> 5, lane = t & 31, l = lane & 15, hi = lane >> 4;
    int b = blockIdx.x >> 3, h = blockIdx.x & 7;
    int q0 = blockIdx.y * 64;
    const __hip_bfloat16* Qp = Qg + (size_t)b * NTOK * ldq + h * DHEAD;
    const __hip_bfloat16* Kp = Kg + (size_t)b * NTOK * ldk + h * DHEAD;
    const __hip_bfloat16* Vp = Vg + (size_t)b * NTOK * ldv + h * DHEAD;

    auto loadK = [&](int buf, int j0) {
#if USE_TDM
        if (wave == 0)
            tdm_load_2d((unsigned)(size_t)&Ks[buf][0][0], Kp + (size_t)j0 * ldk, 64, 64,
                        (unsigned long long)ldk);
#else
#pragma unroll
        for (int i = 0; i < 4; ++i) {
            int c = t * 4 + i, row = c >> 3, c8 = (c & 7) * 8;
            *(uint4*)&Ks[buf][row][c8] = *(const uint4*)(Kp + (size_t)(j0 + row) * ldk + c8);
        }
#endif
    };
    auto loadV = [&](int buf, int j0) {
#pragma unroll
        for (int i = 0; i < 4; ++i) {
            int c = t * 4 + i, row = c >> 3, c8 = (c & 7) * 8;
            uint4 vv = *(const uint4*)(Vp + (size_t)(j0 + row) * ldv + c8);
            const __hip_bfloat16* vp = (const __hip_bfloat16*)&vv;
#pragma unroll
            for (int jj = 0; jj < 8; ++jj) Vs[buf][c8 + jj][row] = vp[jj];
        }
    };

    // ---- prologue: Q tile + K/V chunk 0
#if USE_TDM
    if (wave == 0)
        tdm_load_2d((unsigned)(size_t)&Qs[0][0], Qp + (size_t)q0 * ldq, 64, 64,
                    (unsigned long long)ldq);
#else
#pragma unroll
    for (int i = 0; i < 4; ++i) {
        int c = t * 4 + i, row = c >> 3, c8 = (c & 7) * 8;
        *(uint4*)&Qs[row][c8] = *(const uint4*)(Qp + (size_t)(q0 + row) * ldq + c8);
    }
#endif
    loadK(0, 0);
    loadV(0, 0);
    TDM_WAIT();
    __syncthreads();
    Frag qf[2];
#pragma unroll
    for (int kc = 0; kc < 2; ++kc) {
        int row = wave * 16 + l;
        qf[kc].u[0] = *(const uint4*)&Qs[row][kc * 32 + hi * 8];
        qf[kc].u[1] = *(const uint4*)&Qs[row][kc * 32 + 16 + hi * 8];
    }
    v8f zero = {0.f, 0.f, 0.f, 0.f, 0.f, 0.f, 0.f, 0.f};
    v8f o[4]; float mrow[8], lrow[8];
#pragma unroll
    for (int i = 0; i < 4; ++i) o[i] = zero;
#pragma unroll
    for (int r = 0; r < 8; ++r) { mrow[r] = -1e30f; lrow[r] = 0.f; }

    for (int jc = 0; jc < 16; ++jc) {
        int cur = jc & 1, nxt = cur ^ 1;
        if (jc + 1 < 16) {                   // prefetch next K/V chunk behind compute
            loadK(nxt, (jc + 1) * 64);
            loadV(nxt, (jc + 1) * 64);
        }
        // S = Q K^T  (4 key sub-tiles of 16)
        v8f s[4];
#pragma unroll
        for (int nt = 0; nt < 4; ++nt) {
            Frag k0f, k1f;
            k0f.u[0] = *(const uint4*)&Ks[cur][nt * 16 + l][hi * 8];
            k0f.u[1] = *(const uint4*)&Ks[cur][nt * 16 + l][16 + hi * 8];
            k1f.u[0] = *(const uint4*)&Ks[cur][nt * 16 + l][32 + hi * 8];
            k1f.u[1] = *(const uint4*)&Ks[cur][nt * 16 + l][48 + hi * 8];
            v8f z = __builtin_amdgcn_wmma_f32_16x16x32_bf16(false, qf[0].v, false, k0f.v, (short)0, zero, false, false);
            s[nt] = __builtin_amdgcn_wmma_f32_16x16x32_bf16(false, qf[1].v, false, k1f.v, (short)0, z, false, false);
        }
        // online softmax (rows = hi*8 + r, cols across 16 lanes of each half)
#pragma unroll
        for (int r = 0; r < 8; ++r) {
#pragma unroll
            for (int nt = 0; nt < 4; ++nt) s[nt][r] *= scale;
            float mx = fmaxf(fmaxf(s[0][r], s[1][r]), fmaxf(s[2][r], s[3][r]));
#pragma unroll
            for (int off = 8; off > 0; off >>= 1) mx = fmaxf(mx, __shfl_xor(mx, off, 32));
            float mnew = fmaxf(mrow[r], mx);
            float alpha = __expf(mrow[r] - mnew);
            lrow[r] *= alpha;
#pragma unroll
            for (int d = 0; d < 4; ++d) o[d][r] *= alpha;
            float psum = 0.f;
#pragma unroll
            for (int nt = 0; nt < 4; ++nt) {
                float p = __expf(s[nt][r] - mnew);
                s[nt][r] = p; psum += p;
            }
#pragma unroll
            for (int off = 8; off > 0; off >>= 1) psum += __shfl_xor(psum, off, 32);
            lrow[r] += psum; mrow[r] = mnew;
        }
        // C-layout f32 -> A-layout bf16 via wave-private LDS
#pragma unroll
        for (int nt = 0; nt < 4; ++nt)
#pragma unroll
            for (int r = 0; r < 8; ++r)
                Ps[wave][hi * 8 + r][nt * 16 + l] = __float2bfloat16(s[nt][r]);
        asm volatile("s_wait_dscnt 0x0" ::: "memory");
        Frag pf[2];
#pragma unroll
        for (int kc = 0; kc < 2; ++kc) {
            pf[kc].u[0] = *(const uint4*)&Ps[wave][l][kc * 32 + hi * 8];
            pf[kc].u[1] = *(const uint4*)&Ps[wave][l][kc * 32 + 16 + hi * 8];
        }
#pragma unroll
        for (int d = 0; d < 4; ++d) {
            Frag v0f, v1f;
            v0f.u[0] = *(const uint4*)&Vs[cur][d * 16 + l][hi * 8];
            v0f.u[1] = *(const uint4*)&Vs[cur][d * 16 + l][16 + hi * 8];
            v1f.u[0] = *(const uint4*)&Vs[cur][d * 16 + l][32 + hi * 8];
            v1f.u[1] = *(const uint4*)&Vs[cur][d * 16 + l][48 + hi * 8];
            v8f z = __builtin_amdgcn_wmma_f32_16x16x32_bf16(false, pf[0].v, false, v0f.v, (short)0, o[d], false, false);
            o[d] = __builtin_amdgcn_wmma_f32_16x16x32_bf16(false, pf[1].v, false, v1f.v, (short)0, z, false, false);
        }
        TDM_WAIT();
        __syncthreads();
    }
#pragma unroll
    for (int r = 0; r < 8; ++r) {
        float inv = 1.f / lrow[r];
        int row = q0 + wave * 16 + hi * 8 + r;
#pragma unroll
        for (int d = 0; d < 4; ++d)
            Og[((size_t)b * NTOK + row) * INNER + h * DHEAD + d * 16 + l] =
                __float2bfloat16(o[d][r] * inv);
    }
}

// ------------------------------------------------------------- channel attention
__global__ __launch_bounds__(128) void k_ca_pool(const __hip_bfloat16* __restrict__ t,
                                                 float* __restrict__ pooled) {
    int b = blockIdx.x, c = blockIdx.y * 128 + threadIdx.x;
    const __hip_bfloat16* p = t + (size_t)b * NTOK * INNER + c;
    float mx = -1e30f, sm = 0.f;
    for (int i = 0; i < NTOK; ++i) {
        float v = __bfloat162float(p[(size_t)i * INNER]);
        mx = fmaxf(mx, v); sm += v;
    }
    pooled[b * INNER + c] = mx + sm * (1.f / NTOK);
}

__global__ __launch_bounds__(512) void k_ca_gate(const float* __restrict__ pooled,
                                                 const float* __restrict__ w1,
                                                 const float* __restrict__ w2,
                                                 float* __restrict__ sg) {
    int b = blockIdx.x, t = threadIdx.x;
    __shared__ float hsh[RED], psh[INNER];
    psh[t] = pooled[b * INNER + t];
    __syncthreads();
    if (t < RED) {
        float a = 0.f;
        for (int i = 0; i < INNER; ++i) a += psh[i] * w1[i * RED + t];
        hsh[t] = fmaxf(a, 0.f);
    }
    __syncthreads();
    float a = 0.f;
#pragma unroll
    for (int j = 0; j < RED; ++j) a += hsh[j] * w2[j * INNER + t];
    sg[b * INNER + t] = 1.f / (1.f + __expf(-a));
}

__global__ void k_ca_scale(__hip_bfloat16* __restrict__ t, const float* __restrict__ sg) {
    long i = (long)blockIdx.x * blockDim.x + threadIdx.x;
    if (i >= (long)BATCH * NTOK * INNER) return;
    int c = (int)(i & (INNER - 1));
    int b = (int)(i >> 19);  // / (1024*512)
    t[i] = __float2bfloat16(__bfloat162float(t[i]) * sg[b * INNER + c]);
}

// ---------------------------------------------------------------- launcher
typedef __hip_bfloat16 bf;

extern "C" void kernel_launch(void* const* d_in, const int* in_sizes, int n_in,
                              void* d_out, int out_size, void* d_ws, size_t ws_size,
                              hipStream_t stream) {
    const float* x_in  = (const float*)d_in[0];
    const float* m_in  = (const float*)d_in[1];
    const float* ln1_g = (const float*)d_in[2];
    const float* ln1_b = (const float*)d_in[3];
    const float* w_qkv = (const float*)d_in[4];
    const float* w_o1  = (const float*)d_in[5];
    const float* b_o1  = (const float*)d_in[6];
    const float* ln2_g = (const float*)d_in[7];
    const float* ln2_b = (const float*)d_in[8];
    const float* w_q   = (const float*)d_in[9];
    const float* w_k   = (const float*)d_in[10];
    const float* w_v   = (const float*)d_in[11];
    const float* ca_w1 = (const float*)d_in[12];
    const float* ca_w2 = (const float*)d_in[13];
    const float* w_o2  = (const float*)d_in[14];
    const float* b_o2  = (const float*)d_in[15];
    const float* ln3_g = (const float*)d_in[16];
    const float* ln3_b = (const float*)d_in[17];
    const float* w_f1  = (const float*)d_in[18];
    const float* b_f1  = (const float*)d_in[19];
    const float* w_f2  = (const float*)d_in[20];
    const float* b_f2  = (const float*)d_in[21];

    char* ws = (char*)d_ws;
    size_t off = 0;
    auto alloc = [&](size_t bytes) -> void* {
        void* p = ws + off;
        off = (off + bytes + 255) & ~(size_t)255;
        return p;
    };
    bf* wqkv_b = (bf*)alloc((size_t)DEPTH * DIM * 3 * INNER * 2);
    bf* wo1_b  = (bf*)alloc((size_t)DEPTH * INNER * DIM * 2);
    bf* wq_b   = (bf*)alloc((size_t)DEPTH * DIM * INNER * 2);
    bf* wk_b   = (bf*)alloc((size_t)DEPTH * DIM * INNER * 2);
    bf* wv_b   = (bf*)alloc((size_t)DEPTH * DIM * INNER * 2);
    bf* wo2_b  = (bf*)alloc((size_t)DEPTH * INNER * DIM * 2);
    bf* wf1_b  = (bf*)alloc((size_t)DEPTH * DIM * MLPD * 2);
    bf* wf2_b  = (bf*)alloc((size_t)DEPTH * MLPD * DIM * 2);
    bf* abf1   = (bf*)alloc((size_t)MROWS * DIM * 2);
    bf* abf2   = (bf*)alloc((size_t)MROWS * DIM * 2);
    bf* qkvb   = (bf*)alloc((size_t)MROWS * 3 * INNER * 2);
    bf* obf    = (bf*)alloc((size_t)MROWS * INNER * 2);
    bf* qb     = (bf*)alloc((size_t)MROWS * INNER * 2);
    bf* kb     = (bf*)alloc((size_t)MROWS * INNER * 2);
    bf* vb     = (bf*)alloc((size_t)MROWS * INNER * 2);
    bf* f1b    = (bf*)alloc((size_t)MROWS * MLPD * 2);
    float* pooled = (float*)alloc((size_t)BATCH * INNER * 4);
    float* sgate  = (float*)alloc((size_t)BATCH * INNER * 4);

    auto cvt = [&](const float* s, bf* d, long n) {
        k_cvt<<<dim3((unsigned)((n + 255) / 256)), 256, 0, stream>>>(s, d, n);
    };
    cvt(w_qkv, wqkv_b, (long)DEPTH * DIM * 3 * INNER);
    cvt(w_o1,  wo1_b,  (long)DEPTH * INNER * DIM);
    cvt(w_q,   wq_b,   (long)DEPTH * DIM * INNER);
    cvt(w_k,   wk_b,   (long)DEPTH * DIM * INNER);
    cvt(w_v,   wv_b,   (long)DEPTH * DIM * INNER);
    cvt(w_o2,  wo2_b,  (long)DEPTH * INNER * DIM);
    cvt(w_f1,  wf1_b,  (long)DEPTH * DIM * MLPD);
    cvt(w_f2,  wf2_b,  (long)DEPTH * MLPD * DIM);

    float* xbuf = (float*)d_out;
    long xtot = (long)MROWS * DIM;
    k_copy<<<dim3((unsigned)((xtot + 255) / 256)), 256, 0, stream>>>(x_in, xbuf, xtot);

    dim3 gGemmQKV(MROWS / BM, 3 * INNER / BN);
    dim3 gGemm512(MROWS / BM, DIM / BN);
    dim3 gGemmMLP(MROWS / BM, MLPD / BN);
    dim3 gAttn(BATCH * HEADS, NTOK / 64);
    long scN = (long)BATCH * NTOK * INNER;
    dim3 gScale((unsigned)((scN + 255) / 256));

    for (int i = 0; i < DEPTH; ++i) {
        const bf* Wqkv = wqkv_b + (size_t)i * DIM * 3 * INNER;
        const bf* Wo1  = wo1_b  + (size_t)i * INNER * DIM;
        const bf* Wq   = wq_b   + (size_t)i * DIM * INNER;
        const bf* Wk   = wk_b   + (size_t)i * DIM * INNER;
        const bf* Wv   = wv_b   + (size_t)i * DIM * INNER;
        const bf* Wo2  = wo2_b  + (size_t)i * INNER * DIM;
        const bf* Wf1  = wf1_b  + (size_t)i * DIM * MLPD;
        const bf* Wf2  = wf2_b  + (size_t)i * MLPD * DIM;

        // ---- self-attention block
        k_ln<<<MROWS, 256, 0, stream>>>(xbuf, ln1_g + i * DIM, ln1_b + i * DIM, abf1);
        k_gemm<0><<<gGemmQKV, 256, 0, stream>>>(abf1, Wqkv, nullptr, nullptr, nullptr, qkvb,
                                                MROWS, 3 * INNER, DIM);
        k_attn<<<gAttn, 128, 0, stream>>>(qkvb, qkvb + INNER, qkvb + 2 * INNER, obf,
                                          3 * INNER, 3 * INNER, 3 * INNER, SCALE);
        k_gemm<13><<<gGemm512, 256, 0, stream>>>(obf, Wo1, b_o1 + i * DIM, xbuf, xbuf, nullptr,
                                                 MROWS, DIM, INNER);

        // ---- mixed cross-attention with channel gating
        k_ln<<<MROWS, 256, 0, stream>>>(xbuf, ln2_g + i * DIM, ln2_b + i * DIM, abf1);
        k_ln<<<MROWS, 256, 0, stream>>>(m_in, ln2_g + i * DIM, ln2_b + i * DIM, abf2);
        k_gemm<0><<<gGemm512, 256, 0, stream>>>(abf1, Wq, nullptr, nullptr, nullptr, qb,
                                                MROWS, INNER, DIM);
        k_gemm<0><<<gGemm512, 256, 0, stream>>>(abf2, Wk, nullptr, nullptr, nullptr, kb,
                                                MROWS, INNER, DIM);
        k_gemm<0><<<gGemm512, 256, 0, stream>>>(abf2, Wv, nullptr, nullptr, nullptr, vb,
                                                MROWS, INNER, DIM);
        const float* cw1 = ca_w1 + (size_t)i * INNER * RED;
        const float* cw2 = ca_w2 + (size_t)i * RED * INNER;
        bf* bufs[3] = {qb, kb, vb};
        for (int j = 0; j < 3; ++j) {
            k_ca_pool<<<dim3(BATCH, INNER / 128), 128, 0, stream>>>(bufs[j], pooled);
            k_ca_gate<<<BATCH, INNER, 0, stream>>>(pooled, cw1, cw2, sgate);
            k_ca_scale<<<gScale, 256, 0, stream>>>(bufs[j], sgate);
        }
        k_attn<<<gAttn, 128, 0, stream>>>(qb, kb, vb, obf, INNER, INNER, INNER, SCALE);
        k_gemm<13><<<gGemm512, 256, 0, stream>>>(obf, Wo2, b_o2 + i * DIM, xbuf, xbuf, nullptr,
                                                 MROWS, DIM, INNER);

        // ---- feed-forward
        k_ln<<<MROWS, 256, 0, stream>>>(xbuf, ln3_g + i * DIM, ln3_b + i * DIM, abf1);
        k_gemm<3><<<gGemmMLP, 256, 0, stream>>>(abf1, Wf1, b_f1 + i * MLPD, nullptr, nullptr, f1b,
                                                MROWS, MLPD, DIM);
        k_gemm<13><<<gGemm512, 256, 0, stream>>>(f1b, Wf2, b_f2 + i * DIM, xbuf, xbuf, nullptr,
                                                 MROWS, DIM, MLPD);
    }
    (void)in_sizes; (void)n_in; (void)out_size; (void)ws_size;
}
